// CosSimRouter_expand_49778670960795
// MI455X (gfx1250) — compile-verified, matrix operand order checked by reading, not run on hardware
//
#include <hip/hip_runtime.h>
#include <math.h>

typedef __attribute__((ext_vector_type(16))) _Float16 v16h;
typedef __attribute__((ext_vector_type(8)))  float    v8f;
typedef __attribute__((ext_vector_type(4)))  unsigned int u32x4;
typedef __attribute__((ext_vector_type(8)))  int      i32x8;
typedef __attribute__((ext_vector_type(4)))  int      i32x4;

#define LV   8192
#define LT   512
#define DIM  1024
#define NTILE_V (LV/16)   // 512
#define NTILE_T (LT/16)   // 32

// LDS tile layout: 16 rows x DIM halves with a 16-half (32B) pad after every 512
// halves (TDM pad_interval=256 DWORDs, pad_amount=8 DWORDs) -> row stride 1056.
#define TROW 1056
#define TPAD(kh) (((kh) >= 512) ? 16 : 0)

// ---------------------------------------------------------------- utilities
__device__ __forceinline__ float blockReduceMax(float v, float* red) {
    int tid = threadIdx.x;
    red[tid] = v; __syncthreads();
    for (int s = blockDim.x >> 1; s > 0; s >>= 1) {
        if (tid < s) red[tid] = fmaxf(red[tid], red[tid + s]);
        __syncthreads();
    }
    float r = red[0]; __syncthreads();
    return r;
}
__device__ __forceinline__ float blockReduceSum(float v, float* red) {
    int tid = threadIdx.x;
    red[tid] = v; __syncthreads();
    for (int s = blockDim.x >> 1; s > 0; s >>= 1) {
        if (tid < s) red[tid] += red[tid + s];
        __syncthreads();
    }
    float r = red[0]; __syncthreads();
    return r;
}

__device__ __forceinline__ void wait_tensorcnt0() {
#if __has_builtin(__builtin_amdgcn_s_wait_tensorcnt)
    __builtin_amdgcn_s_wait_tensorcnt((short)0);
#else
    asm volatile("s_wait_tensorcnt 0x0" ::: "memory");
#endif
}

// TDM: DMA one 16 x DIM f16 tile (row-major, stride DIM) from global into LDS,
// with 8-DWORD LDS padding every 256 DWORDs. Issued by the calling wave.
__device__ __forceinline__ void tdm_load_tile(const _Float16* gsrc, void* lds_dst) {
#if __has_builtin(__builtin_amdgcn_tensor_load_to_lds)
    unsigned long long ga = (unsigned long long)(const void*)gsrc;
    unsigned lds_addr = (unsigned)(unsigned long long)lds_dst;  // low 32 bits = LDS offset

    u32x4 g0;
    g0[0] = 1u;                                     // count=1, user descriptor
    g0[1] = lds_addr;                               // lds_addr [63:32]
    g0[2] = (unsigned)(ga & 0xffffffffu);           // global_addr lo
    g0[3] = (unsigned)((ga >> 32) & 0x01ffffffu)    // global_addr hi (57-bit)
          | 0x80000000u;                            // type=2 ("image") [127:126]

    i32x8 g1;
    g1[0] = (1 << 16)                               // data_size=1 -> 2 bytes
          | (1 << 20)                               // pad_enable
          | (7 << 22)                               // pad_interval: 256 DWORDs
          | (7 << 25);                              // pad_amount: 8 DWORDs (32B)
    g1[1] = (DIM & 0xffff) << 16;                   // tensor_dim0 lo16 at [63:48]
    g1[2] = ((DIM >> 16) & 0xffff)                  // tensor_dim0 hi16
          | ((LV & 0xffff) << 16);                  // tensor_dim1 lo16
    g1[3] = ((LV >> 16) & 0xffff)                   // tensor_dim1 hi16
          | ((DIM & 0xffff) << 16);                 // tile_dim0 = DIM
    g1[4] = 16;                                     // tile_dim1 = 16 rows (tile_dim2=0)
    g1[5] = DIM;                                    // tensor_dim0_stride lo32
    g1[6] = 0;                                      // stride0 hi / stride1 lo
    g1[7] = 0;

    i32x4 g2 = {0, 0, 0, 0};
    i32x4 g3 = {0, 0, 0, 0};
#if __clang_major__ >= 23
    i32x8 g4 = {0, 0, 0, 0, 0, 0, 0, 0};
    __builtin_amdgcn_tensor_load_to_lds(g0, g1, g2, g3, g4, 0);
#else
    __builtin_amdgcn_tensor_load_to_lds(g0, g1, g2, g3, 0);
#endif
    wait_tensorcnt0();
#else
    (void)gsrc; (void)lds_dst;
#endif
}

// Cooperative fallback copy with the identical padded LDS layout.
__device__ __forceinline__ void manual_load_tile(const _Float16* gsrc, _Float16* s_a, int tid) {
    for (int e = tid; e < 16 * (DIM / 8); e += 256) {   // 16B chunks
        int r  = e >> 7;                                // DIM/8 = 128 chunks per row
        int kh = (e & 127) * 8;
        const uint4* src = (const uint4*)(gsrc + (size_t)r * DIM + kh);
        uint4* dst = (uint4*)(s_a + (size_t)r * TROW + kh + TPAD(kh));
        *dst = *src;
    }
}

// ---------------------------------------------------------------- K0: init
__global__ void k_zero(float* __restrict__ out, int* __restrict__ counters) {
    size_t idx = (size_t)blockIdx.x * blockDim.x + threadIdx.x;
    float4* o = (float4*)out;
    if (idx < (size_t)LV * DIM / 4) o[idx] = make_float4(0.f, 0.f, 0.f, 0.f);
    if (idx == 0) { counters[0] = 0; counters[1] = 0; }
}

// ---------------------------------------------------------------- K1: normalize -> f16
__global__ void k_normalize(const float* __restrict__ vis, const float* __restrict__ txt,
                            _Float16* __restrict__ vnh, _Float16* __restrict__ tnh) {
    int row = blockIdx.x;
    const float* src; _Float16* dst;
    if (row < LV) { src = vis + (size_t)row * DIM;        dst = vnh + (size_t)row * DIM; }
    else          { src = txt + (size_t)(row - LV) * DIM; dst = tnh + (size_t)(row - LV) * DIM; }
    __shared__ float red[256];
    float ss = 0.f;
    for (int i = threadIdx.x; i < DIM; i += 256) { float v = src[i]; ss += v * v; }
    float tot = blockReduceSum(ss, red);
    float inv = 1.0f / fmaxf(sqrtf(tot), 1e-8f);
    for (int i = threadIdx.x; i < DIM; i += 256) dst[i] = (_Float16)(src[i] * inv);
}

// ---------------------------------------------------------------- K2: stage-1 WMMA rowmax
__global__ void k_stage1(const _Float16* __restrict__ vnh, const _Float16* __restrict__ tnh,
                         const unsigned char* __restrict__ amask, float* __restrict__ rowmax) {
    int strip = blockIdx.x;
    int wave = threadIdx.x >> 5, lane = threadIdx.x & 31;
    int half = lane >> 4, l16 = lane & 15;
    int arow = strip * 16 + l16;
    const _Float16* abase = vnh + (size_t)arow * DIM + half * 16;

    __shared__ float smax[8][16];
    float rmax[8];
#pragma unroll
    for (int r = 0; r < 8; r++) rmax[r] = -3.4e38f;

    for (int ct = wave; ct < NTILE_T; ct += 8) {
        int col = ct * 16 + l16;
        const _Float16* bbase = tnh + (size_t)col * DIM + half * 16;
        v8f c = {};
#pragma unroll 4
        for (int k = 0; k < DIM; k += 32) {
            v16h a = *(const v16h*)(abase + k);
            v16h b = *(const v16h*)(bbase + k);
            c = __builtin_amdgcn_wmma_f32_16x16x32_f16(false, a, false, b, (short)0, c, false, false);
        }
        float mf = amask[col] ? 1.0f : 0.0f;   // where(mask, cos, 0)
#pragma unroll
        for (int r = 0; r < 8; r++) {
            float v = c[r] * mf;               // row = r+8*half, col = l16 (tile-local)
            for (int off = 1; off < 16; off <<= 1)
                v = fmaxf(v, __shfl_xor(v, off, 32));
            rmax[r] = fmaxf(rmax[r], v);
        }
    }
    if (l16 == 0) {
#pragma unroll
        for (int r = 0; r < 8; r++) smax[wave][r + 8 * half] = rmax[r];
    }
    __syncthreads();
    if (threadIdx.x < 16) {
        float m = smax[0][threadIdx.x];
        for (int w = 1; w < 8; w++) m = fmaxf(m, smax[w][threadIdx.x]);
        rowmax[strip * 16 + threadIdx.x] = m;
    }
}

// ---------------------------------------------------------------- K3: double softmax (1 block, 1024 thr)
__global__ void k_softmax2(const float* __restrict__ rowmax,
                           float* __restrict__ scores, float* __restrict__ escore) {
    __shared__ float red[1024];
    int tid = threadIdx.x;
    float v[8];
#pragma unroll
    for (int k = 0; k < 8; k++) v[k] = rowmax[tid + k * 1024] * 10.0f;     // /TEMPERATURE
    float m = v[0];
#pragma unroll
    for (int k = 1; k < 8; k++) m = fmaxf(m, v[k]);
    float M1 = blockReduceMax(m, red);
    float e[8], s = 0.f;
#pragma unroll
    for (int k = 0; k < 8; k++) { e[k] = __expf(v[k] - M1); s += e[k]; }
    float S1 = blockReduceSum(s, red);
    float sc[8], z[8];
#pragma unroll
    for (int k = 0; k < 8; k++) { sc[k] = e[k] / S1; z[k] = sc[k] * 100.0f; }  // /EXPAND_TEMP
    float m2 = z[0];
#pragma unroll
    for (int k = 1; k < 8; k++) m2 = fmaxf(m2, z[k]);
    float M2 = blockReduceMax(m2, red);
    float e2[8], s2 = 0.f;
#pragma unroll
    for (int k = 0; k < 8; k++) { e2[k] = __expf(z[k] - M2); s2 += e2[k]; }
    float S2 = blockReduceSum(s2, red);
#pragma unroll
    for (int k = 0; k < 8; k++) {
        scores[tid + k * 1024] = sc[k];
        escore[tid + k * 1024] = e2[k] / S2;
    }
}

// ---------------------------------------------------------------- K4: stage-1 top-p selection
__global__ void k_select1(const float* __restrict__ scores, int* __restrict__ sel,
                          int* __restrict__ rank1, int* __restrict__ counters) {
    __shared__ float s_sc[LV];
    for (int i = threadIdx.x; i < LV; i += blockDim.x) s_sc[i] = scores[i];
    __syncthreads();
    int i = blockIdx.x * blockDim.x + threadIdx.x;
    float si = s_sc[i];
    float pref = 0.f; int rk = 0;
    for (int j = 0; j < LV; j++) {
        float sj = s_sc[j];
        bool before = (sj > si) || (sj == si && j < i);   // stable desc sort
        rk += before ? 1 : 0;
        pref += (before || j == i) ? sj : 0.f;            // inclusive cumsum at own rank
    }
    int selected = (pref <= 0.3f) ? 1 : 0;                // INITIAL_GAMMA
    sel[i] = selected; rank1[i] = rk;
    if (selected) atomicAdd(&counters[0], 1);
}

// ---------------------------------------------------------------- K5: stage-2 WMMA + online flat softmax
// 64 blocks x 8 waves; wave owns 16 output columns. A tiles staged in LDS via TDM.
__global__ void k_stage2(const _Float16* __restrict__ vnh, const float* __restrict__ escore,
                         const int* __restrict__ sel,
                         float* __restrict__ colmax, float* __restrict__ colsum) {
    __shared__ float         s_es[LV];
    __shared__ unsigned char s_sel[LV];
    __shared__ unsigned char s_tile[NTILE_V];
    __shared__ _Float16      s_a[16 * TROW];            // padded 16 x DIM tile (~33KB)

    for (int i = threadIdx.x; i < LV; i += 256) {
        s_es[i] = escore[i];
        s_sel[i] = (unsigned char)sel[i];
    }
    __syncthreads();
    for (int t = threadIdx.x; t < NTILE_V; t += 256) {
        unsigned char any = 0;
        for (int r = 0; r < 16; r++) any |= s_sel[t * 16 + r];
        s_tile[t] = any;
    }
    __syncthreads();

    int wave = threadIdx.x >> 5, lane = threadIdx.x & 31;
    int half = lane >> 4, l16 = lane & 15;
    int col = blockIdx.x * 128 + wave * 16 + l16;        // B column = vn row j
    const _Float16* bbase = vnh + (size_t)col * DIM + half * 16;

    float runM = -3.4e38f, runS = 0.f;
    for (int t = 0; t < NTILE_V; t++) {
        if (!s_tile[t]) continue;                        // block-uniform skip
        __syncthreads();                                 // s_a readers done before overwrite
#if __has_builtin(__builtin_amdgcn_tensor_load_to_lds)
        if (threadIdx.x < 32)                            // wave 0 issues the TDM op + waits
            tdm_load_tile(vnh + (size_t)t * 16 * DIM, (void*)s_a);
#else
        manual_load_tile(vnh + (size_t)t * 16 * DIM, s_a, threadIdx.x);
#endif
        __syncthreads();                                 // publish tile to all waves

        v8f c = {};
#pragma unroll 4
        for (int k = 0; k < DIM; k += 32) {
            int kh = k + half * 16;
            v16h a = *(const v16h*)(s_a + (size_t)l16 * TROW + kh + TPAD(kh));
            v16h b = *(const v16h*)(bbase + k);
            c = __builtin_amdgcn_wmma_f32_16x16x32_f16(false, a, false, b, (short)0, c, false, false);
        }
#pragma unroll
        for (int r = 0; r < 8; r++) {
            int row = t * 16 + r + 8 * half;
            if (!s_sel[row]) continue;                   // unselected rows -> -inf -> exp 0
            float v = s_es[row] * c[r];
            if (v > runM) { runS = runS * __expf(runM - v) + 1.0f; runM = v; }
            else          { runS += __expf(v - runM); }
        }
    }
    // combine the two half-wave partials that share a column
    float om = __shfl_xor(runM, 16, 32);
    float os = __shfl_xor(runS, 16, 32);
    float m = fmaxf(runM, om);
    float s = 0.f;
    if (runS > 0.f) s += runS * __expf(runM - m);
    if (os  > 0.f)  s += os  * __expf(om  - m);
    if (half == 0) { colmax[col] = m; colsum[col] = s; }
}

// ---------------------------------------------------------------- K6: finalize flat softmax -> agg
__global__ void k_finalize(const float* __restrict__ colmax, const float* __restrict__ colsum,
                           float* __restrict__ agg) {
    __shared__ float red[1024];
    int tid = threadIdx.x;
    float cm[8], cs[8];
#pragma unroll
    for (int k = 0; k < 8; k++) { cm[k] = colmax[tid + k * 1024]; cs[k] = colsum[tid + k * 1024]; }
    float m = cm[0];
#pragma unroll
    for (int k = 1; k < 8; k++) m = fmaxf(m, cm[k]);
    float M = blockReduceMax(m, red);
    float contrib[8], s = 0.f;
#pragma unroll
    for (int k = 0; k < 8; k++) {
        contrib[k] = (cs[k] > 0.f) ? cs[k] * __expf(cm[k] - M) : 0.f;
        s += contrib[k];
    }
    float S = blockReduceSum(s, red);
#pragma unroll
    for (int k = 0; k < 8; k++) agg[tid + k * 1024] = contrib[k] / S;
}

// ---------------------------------------------------------------- K7a: stage-2 candidates
__global__ void k_select2a(const float* __restrict__ agg, const int* __restrict__ sel,
                           int* __restrict__ cand) {
    __shared__ float s_a[LV];
    for (int i = threadIdx.x; i < LV; i += blockDim.x) s_a[i] = agg[i];
    __syncthreads();
    int i = blockIdx.x * blockDim.x + threadIdx.x;
    float ai = s_a[i];
    float pref = 0.f;
    for (int j = 0; j < LV; j++) {
        float aj = s_a[j];
        bool before = (aj > ai) || (aj == ai && j < i);
        pref += (before || j == i) ? aj : 0.f;
    }
    cand[i] = ((pref <= 0.1f) && !sel[i]) ? 1 : 0;        // EXPAND_GAMMA + set-difference
}

// ---------------------------------------------------------------- K7b: rank among candidates
__global__ void k_select2b(const float* __restrict__ agg, const int* __restrict__ cand,
                           int* __restrict__ rank2, int* __restrict__ counters) {
    __shared__ float         s_a[LV];
    __shared__ unsigned char s_c[LV];
    for (int i = threadIdx.x; i < LV; i += blockDim.x) {
        s_a[i] = agg[i];
        s_c[i] = (unsigned char)cand[i];
    }
    __syncthreads();
    int i = blockIdx.x * blockDim.x + threadIdx.x;
    float ai = s_a[i];
    int rk = 0;
    for (int j = 0; j < LV; j++) {
        float aj = s_a[j];
        if (s_c[j] && ((aj > ai) || (aj == ai && j < i))) rk++;
    }
    rank2[i] = rk;
    if (cand[i]) atomicAdd(&counters[1], 1);
}

// ---------------------------------------------------------------- K8: gather/scatter final rows
__global__ void k_scatter(const float* __restrict__ vis, const int* __restrict__ sel,
                          const int* __restrict__ rank1, const int* __restrict__ cand,
                          const int* __restrict__ rank2, const int* __restrict__ counters,
                          float* __restrict__ out) {
    int i = blockIdx.x;
    int dst;
    if (sel[i])       dst = rank1[i];
    else if (cand[i]) dst = counters[0] + rank2[i];
    else return;
    const float4* s = (const float4*)(vis + (size_t)i * DIM);
    float4*       d = (float4*)(out + (size_t)dst * DIM);
    for (int k = threadIdx.x; k < DIM / 4; k += 256) d[k] = s[k];
}

// ---------------------------------------------------------------- launcher
extern "C" void kernel_launch(void* const* d_in, const int* in_sizes, int n_in,
                              void* d_out, int out_size, void* d_ws, size_t ws_size,
                              hipStream_t stream) {
    const float*         vis   = (const float*)d_in[0];
    const float*         txt   = (const float*)d_in[1];
    const unsigned char* amask = (const unsigned char*)d_in[2];
    float* out = (float*)d_out;

    char* ws = (char*)d_ws;
    size_t off = 0;
    _Float16* vnh = (_Float16*)(ws + off); off += (size_t)LV * DIM * 2;   // 16 MB
    _Float16* tnh = (_Float16*)(ws + off); off += (size_t)LT * DIM * 2;   //  1 MB
    float* rowmax = (float*)(ws + off); off += LV * 4;
    float* scores = (float*)(ws + off); off += LV * 4;
    float* escore = (float*)(ws + off); off += LV * 4;
    float* colmax = (float*)(ws + off); off += LV * 4;
    float* colsum = (float*)(ws + off); off += LV * 4;
    float* agg    = (float*)(ws + off); off += LV * 4;
    int* sel   = (int*)(ws + off); off += LV * 4;
    int* rank1 = (int*)(ws + off); off += LV * 4;
    int* cand  = (int*)(ws + off); off += LV * 4;
    int* rank2 = (int*)(ws + off); off += LV * 4;
    int* counters = (int*)(ws + off); off += 256;

    k_zero<<<(LV * DIM / 4 + 255) / 256, 256, 0, stream>>>(out, counters);
    k_normalize<<<LV + LT, 256, 0, stream>>>(vis, txt, vnh, tnh);
    k_stage1<<<NTILE_V, 256, 0, stream>>>(vnh, tnh, amask, rowmax);
    k_softmax2<<<1, 1024, 0, stream>>>(rowmax, scores, escore);
    k_select1<<<LV / 256, 256, 0, stream>>>(scores, sel, rank1, counters);
    k_stage2<<<LV / 128, 256, 0, stream>>>(vnh, escore, sel, colmax, colsum);
    k_finalize<<<1, 1024, 0, stream>>>(colmax, colsum, agg);
    k_select2a<<<LV / 256, 256, 0, stream>>>(agg, sel, cand);
    k_select2b<<<LV / 256, 256, 0, stream>>>(agg, cand, rank2, counters);
    k_scatter<<<LV, 256, 0, stream>>>(vis, sel, rank1, cand, rank2, counters, out);
}
